// VariancePredictor_83648783057120
// MI455X (gfx1250) — matrix operand order repaired and checked by reading, not compiled, not run on hardware
//
#include <hip/hip_runtime.h>
#include <hip/hip_bf16.h>
#include <cstdint>

// Dims from the reference
#define BD   64
#define TD   768
#define CONDD 512
#define INPD 16
#define LATD 64
#define HD   512
#define G4D  2048   // 4*H

// ---------- WMMA plumbing (gfx1250, wave32) ----------
typedef __attribute__((ext_vector_type(16))) __bf16 v16bf;
typedef __attribute__((ext_vector_type(8)))  float  v8f;

struct alignas(16) U4 { unsigned int a, b, c, d; };

union Frag16 {
  v16bf v;
  U4 q[2];
  unsigned short s[16];
};

__device__ __forceinline__ unsigned short f2bf(float f) {
  union { float f; unsigned u; } x; x.f = f;
  unsigned r = x.u + 0x7FFFu + ((x.u >> 16) & 1u);
  return (unsigned short)(r >> 16);
}
__device__ __forceinline__ float sigf(float x) { return 1.f / (1.f + __expf(-x)); }

__device__ __forceinline__ v16bf load2q(const unsigned short* p0, const unsigned short* p1) {
  Frag16 u;
  u.q[0] = *(const U4*)p0;
  u.q[1] = *(const U4*)p1;
  return u.v;
}

// A fragment: 16x32 bf16 from row-major A (ld elems). Per ISA layout:
// lanes 0-15: M=lane, K={k0..k0+7, k0+16..k0+23}; lanes 16-31: M=lane-16, K offset +8.
__device__ __forceinline__ v16bf a_frag(const unsigned short* A, int ld, int mtile, int k0, int lane) {
  int m  = mtile * 16 + (lane & 15);
  int kb = k0 + ((lane >> 4) << 3);
  const unsigned short* r = A + (size_t)m * ld + kb;
  return load2q(r, r + 16);
}
// B fragment: 32x16 (KxN) from W stored [N][K] row-major (PyTorch weight layout).
// lanes 0-15: N=lane, K={k0..k0+7, k0+16..k0+23}; lanes 16-31: N=lane-16, K offset +8.
__device__ __forceinline__ v16bf b_frag(const unsigned short* W, int ld, int n0, int k0, int lane) {
  int n = n0 + (lane & 15);
  int k = k0 + ((lane >> 4) << 3);
  const unsigned short* r = W + (size_t)n * ld + k;
  return load2q(r, r + 16);
}
__device__ __forceinline__ v8f wmma_bf16(v16bf a, v16bf b, v8f c) {
  return __builtin_amdgcn_wmma_f32_16x16x32_bf16(false, a, false, b, (short)0, c, false, false);
}
// A fragment built from f32 source with on-the-fly bf16 conversion
__device__ __forceinline__ v16bf a_frag_f32(const float* A, int ld, int mtile, int k0, int lane) {
  int m  = mtile * 16 + (lane & 15);
  int kb = k0 + ((lane >> 4) << 3);
  const float* r = A + (size_t)m * ld + kb;
  Frag16 u;
#pragma unroll
  for (int j = 0; j < 8; ++j) u.s[j] = f2bf(r[j]);
#pragma unroll
  for (int j = 0; j < 8; ++j) u.s[8 + j] = f2bf(r[16 + j]);
  return u.v;
}

// ---------- weight packing: [Wih | Whh] -> bf16 [N][Kpad] ----------
__global__ void pack_w_kernel(unsigned short* dst, const float* w0, const float* w1,
                              int rows, int k0n, int k1n, int kpad) {
  int total = rows * kpad;
  for (int idx = blockIdx.x * 256 + threadIdx.x; idx < total; idx += gridDim.x * 256) {
    int n = idx / kpad, k = idx - n * kpad;
    float v = 0.f;
    if (k < k0n) v = w0[(size_t)n * k0n + k];
    else if (k < k0n + k1n) v = w1[(size_t)n * k1n + (k - k0n)];
    dst[idx] = f2bf(v);
  }
}

__global__ void zero_kernel(float* p, int n) {
  for (int i = blockIdx.x * 256 + threadIdx.x; i < n; i += gridDim.x * 256) p[i] = 0.f;
}

// ---------- per-step [x;h] assembly (bf16); fwd+bwd fused via blockIdx.y ----------
__global__ void asm_enc0_kernel(unsigned short* xhf, unsigned short* xhb,
                                const float* cond, const float* inp,
                                const float* hf, const float* hb, int tf, int tb) {
  const int K = 1056;
  int idx = blockIdx.x * 256 + threadIdx.x;
  if (idx >= BD * K) return;
  int dir = blockIdx.y;
  unsigned short* xh = dir ? xhb : xhf;
  const float* h = dir ? hb : hf;
  int t = dir ? tb : tf;
  int b = idx / K, k = idx - b * K;
  float v = 0.f;
  if (k < 512)       v = cond[(size_t)b * CONDD * TD + (size_t)k * TD + t];
  else if (k < 528)  v = inp[(size_t)b * INPD * TD + (size_t)(k - 512) * TD + t];
  else if (k < 1040) v = h[b * HD + (k - 528)];
  xh[idx] = f2bf(v);
}

__global__ void asm_enc1_kernel(unsigned short* xhf, unsigned short* xhb,
                                const unsigned short* y0,
                                const float* hf, const float* hb, int tf, int tb) {
  const int K = 1536;
  int idx = blockIdx.x * 256 + threadIdx.x;
  if (idx >= BD * K) return;
  int dir = blockIdx.y;
  unsigned short* xh = dir ? xhb : xhf;
  const float* h = dir ? hb : hf;
  int t = dir ? tb : tf;
  int b = idx / K, k = idx - b * K;
  if (k < 1024) xh[idx] = y0[(size_t)t * BD * 1024 + (size_t)b * 1024 + k];
  else          xh[idx] = f2bf(h[b * HD + (k - 1024)]);
}

__global__ void asm_dec0_kernel(unsigned short* xh, const float* cond, const float* inp,
                                const float* seqz, const float* vecz, const float* h, int t) {
  const int K = 1184;
  int idx = blockIdx.x * 256 + threadIdx.x;
  if (idx >= BD * K) return;
  int b = idx / K, k = idx - b * K;
  float v = 0.f;
  if (k < 512)       v = cond[(size_t)b * CONDD * TD + (size_t)k * TD + t];
  else if (k < 528)  v = (t > 0) ? inp[(size_t)b * INPD * TD + (size_t)(k - 512) * TD + (t - 1)] : 0.f;
  else if (k < 592)  v = seqz[(size_t)b * TD * LATD + (size_t)t * LATD + (k - 528)];
  else if (k < 656)  v = vecz[b * LATD + (k - 592)];
  else if (k < 1168) v = h[b * HD + (k - 656)];
  xh[idx] = f2bf(v);
}

__global__ void asm_dec1_kernel(unsigned short* xh, const float* h0, const float* h1) {
  const int K = 1024;
  int idx = blockIdx.x * 256 + threadIdx.x;
  if (idx >= BD * K) return;
  int b = idx / K, k = idx - b * K;
  xh[idx] = f2bf(k < 512 ? h0[b * HD + k] : h1[b * HD + (k - 512)]);
}

// ---------- fused LSTM cell core: gates GEMM (WMMA) + nonlinearity + state update ----------
// blockIdx.x = hidden slice of 32 (16 slices); block = 256 threads = 8 waves.
// Wave w handles gate (w>>1) and N-subtile (w&1) for all 4 M tiles of the batch.
__device__ __forceinline__ void lstm_core(float (*gsm)[64][33],
                                          const unsigned short* __restrict__ xh, int kpad,
                                          const unsigned short* __restrict__ W,
                                          const float* __restrict__ bias,
                                          float* __restrict__ hst, float* __restrict__ cst,
                                          const int* __restrict__ lengths, int t, bool encoder,
                                          unsigned short* __restrict__ ybf, int ycols, int ycol0,
                                          float* __restrict__ finh, float* __restrict__ finc) {
  const int lane = threadIdx.x & 31;
  const int w    = threadIdx.x >> 5;
  const int gq   = w >> 1, nsub = w & 1;
  const int ncol = gq * 512 + blockIdx.x * 32 + nsub * 16;   // gate-matrix column base

  v8f z8 = {0.f, 0.f, 0.f, 0.f, 0.f, 0.f, 0.f, 0.f};
  v8f acc[4] = {z8, z8, z8, z8};
  for (int k0 = 0; k0 < kpad; k0 += 32) {
    v16bf bf = b_frag(W, kpad, ncol, k0, lane);
    __builtin_prefetch(W + (size_t)ncol * kpad + k0 + 256, 0, 1);
#pragma unroll
    for (int mt = 0; mt < 4; ++mt) {
      v16bf af = a_frag(xh, kpad, mt, k0, lane);
      acc[mt] = wmma_bf16(af, bf, acc[mt]);
    }
  }
  // C layout: VGPR r holds M = r + (lane<16 ? 0 : 8), N = lane&15
  const int colw  = nsub * 16 + (lane & 15);
  const int rbase = (lane >> 4) << 3;
#pragma unroll
  for (int mt = 0; mt < 4; ++mt)
#pragma unroll
    for (int r = 0; r < 8; ++r)
      gsm[gq][mt * 16 + rbase + r][colw] = acc[mt][r];
  __syncthreads();

#pragma unroll
  for (int p = 0; p < 8; ++p) {
    int idx = threadIdx.x + p * 256;          // 2048 = 64 batch x 32 cols
    int b = idx >> 5, j = idx & 31;
    int n = blockIdx.x * 32 + j;
    float iv = gsm[0][b][j] + bias[n];
    float fv = gsm[1][b][j] + bias[512 + n];
    float gv = gsm[2][b][j] + bias[1024 + n];
    float ov = gsm[3][b][j] + bias[1536 + n];
    int si = b * HD + n;
    float co = cst[si], ho = hst[si];
    float cn = sigf(fv) * co + sigf(iv) * tanhf(gv);
    float hn = sigf(ov) * tanhf(cn);
    if (encoder) {
      bool valid = t < lengths[b];
      hst[si] = valid ? hn : ho;
      cst[si] = valid ? cn : co;
      ybf[(size_t)t * BD * ycols + (size_t)b * ycols + ycol0 + n] = f2bf(valid ? hn : 0.f);
    } else {
      float h2 = 0.1f * ho + 0.9f * hn;       // eval-mode zoneout blend
      float c2 = 0.1f * co + 0.9f * cn;
      hst[si] = h2; cst[si] = c2;
      if (lengths[b] == t + 1) { finh[si] = h2; finc[si] = c2; }
    }
  }
}

// Encoder step: fwd (blockIdx.y==0) and bwd (==1) cells run in one dispatch —
// they are independent, so fusing them halves the serial depth of the recurrence.
__global__ void lstm_enc_pair_kernel(const unsigned short* __restrict__ xhf,
                                     const unsigned short* __restrict__ xhb, int kpad,
                                     const unsigned short* __restrict__ Wf,
                                     const unsigned short* __restrict__ Wb,
                                     const float* __restrict__ bf_, const float* __restrict__ bb_,
                                     float* hf, float* cf, float* hb, float* cb,
                                     const int* __restrict__ lengths, int tf, int tb,
                                     unsigned short* __restrict__ ybf, int ycols) {
  __shared__ float gsm[4][64][33];
  if (blockIdx.y == 0)
    lstm_core(gsm, xhf, kpad, Wf, bf_, hf, cf, lengths, tf, true, ybf, ycols, 0,   nullptr, nullptr);
  else
    lstm_core(gsm, xhb, kpad, Wb, bb_, hb, cb, lengths, tb, true, ybf, ycols, 512, nullptr, nullptr);
}

// Decoder step (zoneout + final-state capture)
__global__ void lstm_dec_kernel(const unsigned short* __restrict__ xh, int kpad,
                                const unsigned short* __restrict__ W,
                                const float* __restrict__ bias,
                                float* hst, float* cst,
                                const int* __restrict__ lengths, int t,
                                float* finh, float* finc) {
  __shared__ float gsm[4][64][33];
  lstm_core(gsm, xh, kpad, W, bias, hst, cst, lengths, t, false, nullptr, 0, 0, finh, finc);
}

// ---------- per-step decoder projection: [64,512] @ [512,16] ----------
__global__ void proj_step_kernel(const float* __restrict__ h,
                                 const unsigned short* __restrict__ Wp,
                                 const float* __restrict__ pb,
                                 float* __restrict__ pred,
                                 const int* __restrict__ lengths, int t) {
  const int lane = threadIdx.x & 31;
  const int mt   = threadIdx.x >> 5;          // 4 waves, one M tile each
  v8f acc = {0.f, 0.f, 0.f, 0.f, 0.f, 0.f, 0.f, 0.f};
  for (int k0 = 0; k0 < HD; k0 += 32) {
    v16bf bf = b_frag(Wp, HD, 0, k0, lane);
    v16bf af = a_frag_f32(h, HD, mt, k0, lane);
    acc = wmma_bf16(af, bf, acc);
  }
  int i = lane & 15;
  int rbase = (lane >> 4) << 3;
#pragma unroll
  for (int r = 0; r < 8; ++r) {
    int b = mt * 16 + rbase + r;
    float v = acc[r] + pb[i];
    pred[(size_t)b * INPD * TD + (size_t)i * TD + t] = (t < lengths[b]) ? v : 0.f;
  }
}

// ---------- seq posterior: [T*B,1024] @ [1024,128] ----------
__global__ void seq_post_kernel(const unsigned short* __restrict__ y1,
                                const unsigned short* __restrict__ Wsp,
                                const float* __restrict__ bsp,
                                float* __restrict__ seqz, float* __restrict__ seqmu,
                                float* __restrict__ seqlv) {
  const int lane = threadIdx.x & 31;
  const int w    = threadIdx.x >> 5;          // 8 waves = 8 N tiles
  const int n0   = w * 16;
  v8f acc = {0.f, 0.f, 0.f, 0.f, 0.f, 0.f, 0.f, 0.f};
  for (int k0 = 0; k0 < 1024; k0 += 32) {
    v16bf bf = b_frag(Wsp, 1024, n0, k0, lane);
    v16bf af = a_frag(y1, 1024, blockIdx.x, k0, lane);
    acc = wmma_bf16(af, bf, acc);
  }
  int n = n0 + (lane & 15);
  int rbase = (lane >> 4) << 3;
#pragma unroll
  for (int r = 0; r < 8; ++r) {
    int m = blockIdx.x * 16 + rbase + r;      // m = t*64 + b  (y stored [T,B,2H])
    int tt = m >> 6, b = m & 63;
    float v = acc[r] + bsp[n];
    if (n < LATD) {
      size_t o = (size_t)b * TD * LATD + (size_t)tt * LATD + n;
      seqmu[o] = v; seqz[o] = v;
    } else {
      size_t o = (size_t)b * TD * LATD + (size_t)tt * LATD + (n - LATD);
      seqlv[o] = v;
    }
  }
}

// ---------- vec posterior: h_all [64,2048] @ [2048,128] ----------
__global__ void vec_post_kernel(const float* f0, const float* f1, const float* f2, const float* f3,
                                const unsigned short* __restrict__ Wvp,
                                const float* __restrict__ bvp,
                                float* __restrict__ vecz, float* __restrict__ vecmu,
                                float* __restrict__ veclv) {
  const int lane = threadIdx.x & 31;
  const int w    = threadIdx.x >> 5;
  const int n0   = w * 16;
  v8f z8 = {0.f, 0.f, 0.f, 0.f, 0.f, 0.f, 0.f, 0.f};
  v8f acc[4] = {z8, z8, z8, z8};
  for (int k0 = 0; k0 < 2048; k0 += 32) {
    v16bf bf = b_frag(Wvp, 2048, n0, k0, lane);
#pragma unroll
    for (int mt = 0; mt < 4; ++mt) {
      int m  = mt * 16 + (lane & 15);
      int kb = k0 + ((lane >> 4) << 3);
      Frag16 u;
#pragma unroll
      for (int j = 0; j < 16; ++j) {
        int k = kb + (j < 8 ? j : 8 + j);     // matches A-fragment K pattern
        const float* p = (k < 512) ? f0 : (k < 1024) ? f1 : (k < 1536) ? f2 : f3;
        u.s[j] = f2bf(p[m * HD + (k & 511)]);
      }
      acc[mt] = wmma_bf16(u.v, bf, acc[mt]);
    }
  }
  int n = n0 + (lane & 15);
  int rbase = (lane >> 4) << 3;
#pragma unroll
  for (int mt = 0; mt < 4; ++mt)
#pragma unroll
    for (int r = 0; r < 8; ++r) {
      int b = mt * 16 + rbase + r;
      float v = acc[mt][r] + bvp[n];
      if (n < LATD) { vecmu[b * LATD + n] = v; vecz[b * LATD + n] = v; }
      else          veclv[b * LATD + (n - LATD)] = v;
    }
}

__global__ void final_states_kernel(const float* f0h, const float* f0c,
                                    const float* f1h, const float* f1c, float* out) {
  int idx = blockIdx.x * 256 + threadIdx.x;   // [B, 2, 1024]
  if (idx >= BD * 2 * 1024) return;
  int b = idx >> 11, rest = idx & 2047;
  int l = rest >> 10, k = rest & 1023;
  const float* src = (l == 0) ? (k < 512 ? f0h : f0c) : (k < 512 ? f1h : f1c);
  out[idx] = src[b * HD + (k & 511)];
}

// ---------- host driver ----------
extern "C" void kernel_launch(void* const* d_in, const int* in_sizes, int n_in,
                              void* d_out, int out_size, void* d_ws, size_t ws_size,
                              hipStream_t stream) {
  (void)in_sizes; (void)n_in; (void)out_size; (void)ws_size;
  const float* cond = (const float*)d_in[0];
  const float* inp  = (const float*)d_in[1];
  const int*   len  = (const int*)d_in[2];
  const float* P[24];
  for (int i = 0; i < 24; ++i) P[i] = (const float*)d_in[3 + i];
  // params insertion order: enc[0].f{Wih,Whh,b}, enc[0].b{..}, enc[1].f{..}, enc[1].b{..},
  //                         dec[0]{..}, dec[1]{..}, seq_post{W,b}, vec_post{W,b}, proj{W,b}
  const float* e_wih[4] = { P[0], P[3], P[6], P[9]  };
  const float* e_whh[4] = { P[1], P[4], P[7], P[10] };
  const float* e_b[4]   = { P[2], P[5], P[8], P[11] };
  const float* d_wih[2] = { P[12], P[15] };
  const float* d_whh[2] = { P[13], P[16] };
  const float* d_bi[2]  = { P[14], P[17] };
  const float* seqW = P[18]; const float* seqb = P[19];
  const float* vecW = P[20]; const float* vecb = P[21];
  const float* prjW = P[22]; const float* prjb = P[23];

  uint8_t* ws = (uint8_t*)d_ws;
  size_t off = 0;
  auto alloc = [&](size_t bytes) -> void* {
    size_t o = (off + 255) & ~(size_t)255; off = o + bytes; return (void*)(ws + o);
  };

  const int Kenc[4] = {1056, 1056, 1536, 1536};
  const int Kdec[2] = {1184, 1024};
  unsigned short* Wenc[4];
  for (int c = 0; c < 4; ++c) Wenc[c] = (unsigned short*)alloc((size_t)G4D * Kenc[c] * 2);
  unsigned short* Wdec[2];
  for (int c = 0; c < 2; ++c) Wdec[c] = (unsigned short*)alloc((size_t)G4D * Kdec[c] * 2);
  unsigned short* Wprj = (unsigned short*)alloc((size_t)16 * 512 * 2);
  unsigned short* Wseq = (unsigned short*)alloc((size_t)128 * 1024 * 2);
  unsigned short* Wvec = (unsigned short*)alloc((size_t)128 * 2048 * 2);
  unsigned short* y0   = (unsigned short*)alloc((size_t)TD * BD * 1024 * 2);
  unsigned short* y1   = (unsigned short*)alloc((size_t)TD * BD * 1024 * 2);
  float* st = (float*)alloc((size_t)16 * BD * HD * 4);     // 16 state slots
  auto slot = [&](int i) { return st + (size_t)i * BD * HD; };
  float *eh[4], *ec[4];
  for (int c = 0; c < 4; ++c) { eh[c] = slot(c); ec[c] = slot(4 + c); }
  float* dh[2] = { slot(8),  slot(10) };
  float* dc[2] = { slot(9),  slot(11) };
  float* fh[2] = { slot(12), slot(14) };
  float* fc[2] = { slot(13), slot(15) };
  unsigned short* xh[6];
  const int Kx[6] = {1056, 1056, 1536, 1536, 1184, 1024};
  for (int i = 0; i < 6; ++i) xh[i] = (unsigned short*)alloc((size_t)BD * Kx[i] * 2);

  float* out     = (float*)d_out;
  float* o_pred  = out;                          // [64,16,768]
  float* o_fs    = out + 786432;                 // [64,2,1024]
  float* o_seqz  = out + 917504;                 // [64,768,64]
  float* o_seqmu = out + 4063232;
  float* o_seqlv = out + 7208960;
  float* o_vecz  = out + 10354688;               // [64,64]
  float* o_vecmu = out + 10358784;
  float* o_veclv = out + 10362880;

  // 1) convert/pack weights to bf16 combined [Wih|Whh] layout (L2-resident working set)
  for (int c = 0; c < 4; ++c) {
    int kih = (c < 2) ? 528 : 1024;
    pack_w_kernel<<<4096, 256, 0, stream>>>(Wenc[c], e_wih[c], e_whh[c], G4D, kih, HD, Kenc[c]);
  }
  pack_w_kernel<<<4096, 256, 0, stream>>>(Wdec[0], d_wih[0], d_whh[0], G4D, 656, HD, Kdec[0]);
  pack_w_kernel<<<4096, 256, 0, stream>>>(Wdec[1], d_wih[1], d_whh[1], G4D, 512, HD, Kdec[1]);
  pack_w_kernel<<<64,   256, 0, stream>>>(Wprj, prjW, nullptr, 16,  512,  0, 512);
  pack_w_kernel<<<512,  256, 0, stream>>>(Wseq, seqW, nullptr, 128, 1024, 0, 1024);
  pack_w_kernel<<<1024, 256, 0, stream>>>(Wvec, vecW, nullptr, 128, 2048, 0, 2048);
  zero_kernel<<<2048, 256, 0, stream>>>(st, 16 * BD * HD);

  // 2) encoder layer 0: fwd+bwd fused per step (independent cells, one dispatch)
  for (int t = 0; t < TD; ++t) {
    int tr = TD - 1 - t;
    asm_enc0_kernel<<<dim3(264, 2), 256, 0, stream>>>(xh[0], xh[1], cond, inp, eh[0], eh[1], t, tr);
    lstm_enc_pair_kernel<<<dim3(16, 2), 256, 0, stream>>>(xh[0], xh[1], 1056, Wenc[0], Wenc[1],
                                                          e_b[0], e_b[1], eh[0], ec[0], eh[1], ec[1],
                                                          len, t, tr, y0, 1024);
  }
  // 3) encoder layer 1
  for (int t = 0; t < TD; ++t) {
    int tr = TD - 1 - t;
    asm_enc1_kernel<<<dim3(384, 2), 256, 0, stream>>>(xh[2], xh[3], y0, eh[2], eh[3], t, tr);
    lstm_enc_pair_kernel<<<dim3(16, 2), 256, 0, stream>>>(xh[2], xh[3], 1536, Wenc[2], Wenc[3],
                                                          e_b[2], e_b[3], eh[2], ec[2], eh[3], ec[3],
                                                          len, t, tr, y1, 1024);
  }
  // 4) posteriors
  seq_post_kernel<<<3072, 256, 0, stream>>>(y1, Wseq, seqb, o_seqz, o_seqmu, o_seqlv);
  vec_post_kernel<<<1, 256, 0, stream>>>(eh[0], eh[1], eh[2], eh[3], Wvec, vecb,
                                         o_vecz, o_vecmu, o_veclv);
  // 5) decoder (teacher forcing), serial in t and across layers
  for (int t = 0; t < TD; ++t) {
    asm_dec0_kernel<<<296, 256, 0, stream>>>(xh[4], cond, inp, o_seqz, o_vecz, dh[0], t);
    lstm_dec_kernel<<<16, 256, 0, stream>>>(xh[4], 1184, Wdec[0], d_bi[0], dh[0], dc[0],
                                            len, t, fh[0], fc[0]);
    asm_dec1_kernel<<<256, 256, 0, stream>>>(xh[5], dh[0], dh[1]);
    lstm_dec_kernel<<<16, 256, 0, stream>>>(xh[5], 1024, Wdec[1], d_bi[1], dh[1], dc[1],
                                            len, t, fh[1], fc[1]);
    proj_step_kernel<<<1, 128, 0, stream>>>(dh[1], Wprj, prjb, o_pred, len, t);
  }
  // 6) final decoder states
  final_states_kernel<<<512, 256, 0, stream>>>(fh[0], fc[0], fh[1], fc[1], o_fs);
}
